// SpanQualifier_38216619000312
// MI455X (gfx1250) — matrix operand order repaired and checked by reading, not compiled, not run on hardware
//
#include <hip/hip_runtime.h>
#include <hip/hip_bf16.h>

typedef __bf16 bf16;
typedef __attribute__((ext_vector_type(8)))  bf16  v8bf;
typedef __attribute__((ext_vector_type(16))) bf16  v16bf;
typedef __attribute__((ext_vector_type(8)))  float v8f;

constexpr int cL  = 256;
constexpr int cD1 = 768;
constexpr int cD2 = 128;
constexpr int cGAP = 30;
constexpr int cB  = 2;

// ---- CDNA5 async global->LDS copy (16B per lane), tracked by ASYNCcnt ------
__device__ __forceinline__ void async_b128(unsigned lds_off, const void* gaddr) {
    asm volatile("global_load_async_to_lds_b128 %0, %1, off"
                 :: "v"(lds_off), "v"(gaddr) : "memory");
}
__device__ __forceinline__ void wait_async0() {
    asm volatile("s_wait_asynccnt 0" ::: "memory");
}
__device__ __forceinline__ unsigned lds_off32(const void* p) {
    return (unsigned)(unsigned long long)p;   // low 32 bits = LDS byte offset
}

// ---------------------------------------------------------------- fp32 -> bf16
__global__ void cvt_f32_bf16(const float* __restrict__ s, bf16* __restrict__ d, int n) {
    int i = blockIdx.x * blockDim.x + threadIdx.x;
    if (i < n) d[i] = (bf16)s[i];
}

// transpose-convert: d[n][k] = (bf16) s[k][n]   (s is [K][N] fp32)
__global__ void cvt_f32_bf16_t(const float* __restrict__ s, bf16* __restrict__ d,
                               int K, int N) {
    long i = (long)blockIdx.x * blockDim.x + threadIdx.x;
    if (i < (long)K * N) {
        int n = (int)(i / K), k = (int)(i % K);
        d[i] = (bf16)s[(long)k * N + n];
    }
}

// conv weights: s[dd][ci][co] fp32 -> d[dd][co][ci] bf16
__global__ void cvt_convw_t(const float* __restrict__ s, bf16* __restrict__ d) {
    long i = (long)blockIdx.x * blockDim.x + threadIdx.x;
    if (i < 25L * cD2 * cD2) {
        int dd = (int)(i / (cD2 * cD2));
        int r  = (int)(i % (cD2 * cD2));
        int co = r / cD2, ci = r % cD2;
        d[i] = (bf16)s[((long)dd * cD2 + ci) * cD2 + co];
    }
}

// ---------------------------------------------------------------- generic GEMM
// C[M,N] = act(A[M,K] @ B + bias); A bf16 [M,K] row-major, BT bf16 [N][K]
// (pre-transposed). 64x64 tile / 128 threads (4 wave32). Batched via strides.
// transM>0: bf16 output written transposed per batch of transM rows.
__global__ __launch_bounds__(128)
void gemm_bf16(const bf16* __restrict__ A, const bf16* __restrict__ BT,
               const float* __restrict__ bias,
               float* __restrict__ Cf, bf16* __restrict__ Cb,
               int M, int N, int K, int relu, int transM,
               long sA, long sB, long sC)
{
    __shared__ __align__(16) bf16 As[64][32];   // [m][k]
    __shared__ __align__(16) bf16 Bs[64][32];   // [n][k]

    const int tid  = threadIdx.x;
    const int wave = tid >> 5;
    const int lane = tid & 31;
    const int lhalf = lane >> 4;
    const int lmod  = lane & 15;

    A  += (long)blockIdx.z * sA;
    BT += (long)blockIdx.z * sB;
    const long cOff = (long)blockIdx.z * sC;
    const int m0 = blockIdx.x * 64;
    const int n0 = blockIdx.y * 64;

    v8f acc[4] = {};

    const int idx = tid * 16;
    const int r   = idx >> 5, kk = idx & 31;    // kk in {0,16}

    for (int k0 = 0; k0 < K; k0 += 32) {
        {   // A tile 64x32: 16 contiguous bf16 per thread via async b128 x2
            const bf16* g = A + (long)(m0 + r) * K + k0 + kk;
            const unsigned l = lds_off32(&As[r][kk]);
            async_b128(l, g);
            async_b128(l + 16, g + 8);
        }
        {   // B tile (rows of BT): identical pattern
            const bf16* g = BT + (long)(n0 + r) * K + k0 + kk;
            const unsigned l = lds_off32(&Bs[r][kk]);
            async_b128(l, g);
            async_b128(l + 16, g + 8);
        }
        wait_async0();
        __syncthreads();

        const int ar = wave * 16 + lmod;
        union { v16bf v; v8bf h[2]; } a;
        a.h[0] = *(const v8bf*)&As[ar][lhalf * 8];        // K = kbase..kbase+7
        a.h[1] = *(const v8bf*)&As[ar][16 + lhalf * 8];   // K = 16+kbase..
#pragma unroll
        for (int s = 0; s < 4; ++s) {
            const int nc = s * 16 + lmod;
            union { v16bf v; v8bf h[2]; } bb;
            bb.h[0] = *(const v8bf*)&Bs[nc][lhalf * 8];
            bb.h[1] = *(const v8bf*)&Bs[nc][16 + lhalf * 8];
            acc[s] = __builtin_amdgcn_wmma_f32_16x16x32_bf16(
                false, a.v, false, bb.v, (short)0, acc[s], false, false);
        }
        __syncthreads();
    }

#pragma unroll
    for (int s = 0; s < 4; ++s) {
        const int col = n0 + s * 16 + lmod;
        const float bv = bias ? bias[col] : 0.0f;
#pragma unroll
        for (int rr = 0; rr < 8; ++rr) {
            const int row = m0 + wave * 16 + lhalf * 8 + rr;
            float v = acc[s][rr] + bv;
            if (relu) v = fmaxf(v, 0.0f);
            if (Cf) Cf[cOff + (long)row * N + col] = v;
            if (Cb) {
                if (transM > 0) {
                    const int b2 = row / transM, r2 = row % transM;
                    Cb[(long)b2 * sC + (long)col * transM + r2] = (bf16)v;
                } else {
                    Cb[cOff + (long)row * N + col] = (bf16)v;
                }
            }
        }
    }
}

// ---------------------------------------------------------------- 5x5 SAME conv
// Implicit GEMM, 25 accumulated K=128 steps; tile = 64 j-pixels x 128 channels.
// Wt is pre-transposed [dd][co][ci] bf16.
__global__ __launch_bounds__(128)
void conv5x5_bf16(const bf16* __restrict__ X, const bf16* __restrict__ Wt,
                  const float* __restrict__ bias, bf16* __restrict__ Y)
{
    __shared__ __align__(16) bf16 As[64][cD2];     // 16 KB pixels x ci
    __shared__ __align__(16) bf16 Bt[cD2][cD2];    // 32 KB co x ci

    const int tid  = threadIdx.x;
    const int wave = tid >> 5;
    const int lane = tid & 31;
    const int lhalf = lane >> 4;
    const int lmod  = lane & 15;

    const int j0 = blockIdx.x * 64;
    const int i  = blockIdx.y;
    const int b  = blockIdx.z;

    v8f acc[8] = {};

    for (int dd = 0; dd < 25; ++dd) {
        const int di = dd / 5 - 2, dj = dd % 5 - 2;
        const int ii = i + di;
        if (ii < 0 || ii >= cL) continue;          // uniform across block
        // stage shifted activations (zero-pad at j borders)
        {
            const int p  = tid >> 1;
            const int c0 = (tid & 1) * 64;
            const int jj = j0 + p + dj;
            const unsigned l = lds_off32(&As[p][c0]);
            if (jj >= 0 && jj < cL) {              // per-lane EXEC masks asyncs
                const bf16* g = X + (((long)(b * cL + ii)) * cL + jj) * cD2 + c0;
#pragma unroll
                for (int q = 0; q < 8; ++q) async_b128(l + q * 16, g + q * 8);
            } else {
                const uint4 z = make_uint4(0, 0, 0, 0);
#pragma unroll
                for (int q = 0; q < 8; ++q) *(uint4*)&As[p][c0 + q * 8] = z;
            }
        }
        // stage Wt[dd] slice: fully contiguous, 128 bf16 per thread
        {
            const bf16* g = Wt + (long)dd * cD2 * cD2 + tid * cD2;
            const unsigned l = lds_off32(&Bt[0][0]) + tid * (cD2 * 2);
#pragma unroll
            for (int q = 0; q < 16; ++q) async_b128(l + q * 16, g + q * 8);
        }
        wait_async0();
        __syncthreads();

        const int pr = wave * 16 + lmod;
#pragma unroll
        for (int kt = 0; kt < 4; ++kt) {
            union { v16bf v; v8bf h[2]; } a;
            a.h[0] = *(const v8bf*)&As[pr][kt * 32 + lhalf * 8];
            a.h[1] = *(const v8bf*)&As[pr][kt * 32 + 16 + lhalf * 8];
#pragma unroll
            for (int s = 0; s < 8; ++s) {
                union { v16bf v; v8bf h[2]; } bb;
                bb.h[0] = *(const v8bf*)&Bt[s * 16 + lmod][kt * 32 + lhalf * 8];
                bb.h[1] = *(const v8bf*)&Bt[s * 16 + lmod][kt * 32 + 16 + lhalf * 8];
                acc[s] = __builtin_amdgcn_wmma_f32_16x16x32_bf16(
                    false, a.v, false, bb.v, (short)0, acc[s], false, false);
            }
        }
        __syncthreads();
    }

#pragma unroll
    for (int s = 0; s < 8; ++s) {
        const int col = s * 16 + lmod;
        const float bv = bias[col];
#pragma unroll
        for (int rr = 0; rr < 8; ++rr) {
            const int j = j0 + wave * 16 + lhalf * 8 + rr;
            Y[(((long)(b * cL + i)) * cL + j) * cD2 + col] = (bf16)(acc[s][rr] + bv);
        }
    }
}

// ----------------------------------------------- span_enum + LayerNorm (+mask)
__global__ __launch_bounds__(128)
void span_ln(const float* __restrict__ Ps, const float* __restrict__ Pe,
             const float* __restrict__ emb, const float* __restrict__ g,
             const float* __restrict__ bb, const float* __restrict__ masks,
             int applyMask, bf16* __restrict__ Out)
{
    __shared__ float red[cD2];
    const int c = threadIdx.x;
    const int j = blockIdx.x, i = blockIdx.y, b = blockIdx.z;
    const int ad = (i > j) ? (i - j) : (j - i);

    float v = Ps[((long)(b * cL + i)) * cD2 + c]
            + Pe[((long)(b * cL + j)) * cD2 + c]
            + emb[(long)ad * cD2 + c];

    red[c] = v; __syncthreads();
    for (int s = 64; s > 0; s >>= 1) { if (c < s) red[c] += red[c + s]; __syncthreads(); }
    const float mean = red[0] * (1.0f / cD2);
    __syncthreads();

    const float d = v - mean;
    red[c] = d * d; __syncthreads();
    for (int s = 64; s > 0; s >>= 1) { if (c < s) red[c] += red[c + s]; __syncthreads(); }
    const float var = red[0] * (1.0f / cD2);

    float o = d * (1.0f / sqrtf(var + 1e-12f)) * g[c] + bb[c];
    if (applyMask) {
        const float mm = masks[b * cL + i] * masks[b * cL + j] *
                         ((i <= j && (j - i) <= cGAP) ? 1.0f : 0.0f);
        o *= mm;
    }
    Out[(((long)(b * cL + i)) * cL + j) * cD2 + c] = (bf16)o;
}

// ------------------------------------------- scorer: dot(h,w2)+b2 - 1e4*(1-mm)
__global__ __launch_bounds__(128)
void score_dot(const bf16* __restrict__ H, const float* __restrict__ w2,
               const float* __restrict__ b2, const float* __restrict__ masks,
               float* __restrict__ Out)
{
    const int wave = threadIdx.x >> 5, lane = threadIdx.x & 31;
    const long pix = (long)blockIdx.x * 4 + wave;
    const int j = (int)(pix % cL);
    const int i = (int)((pix / cL) % cL);
    const int b = (int)(pix / ((long)cL * cL));
    const bf16* h = H + pix * cD2;

    float s = 0.0f;
#pragma unroll
    for (int q = 0; q < 4; ++q) {
        const int c = lane + q * 32;
        s += (float)h[c] * w2[c];
    }
#pragma unroll
    for (int off = 16; off > 0; off >>= 1) s += __shfl_down(s, off, 32);
    if (lane == 0) {
        const float mm = masks[b * cL + i] * masks[b * cL + j] *
                         ((i <= j && (j - i) <= cGAP) ? 1.0f : 0.0f);
        Out[pix] = s + b2[0] - 10000.0f * (1.0f - mm);
    }
}

// ---------------------------------------------------------------- softmaxes
__global__ __launch_bounds__(256)
void softmax_row(const float* __restrict__ G, bf16* __restrict__ P)
{
    __shared__ float red[cL];
    const int j = threadIdx.x;
    const long row = blockIdx.x;                  // b*L + i
    const float v = G[row * cL + j];
    red[j] = v; __syncthreads();
    for (int s = 128; s > 0; s >>= 1) { if (j < s) red[j] = fmaxf(red[j], red[j + s]); __syncthreads(); }
    const float mx = red[0]; __syncthreads();
    const float e = expf(v - mx);
    red[j] = e; __syncthreads();
    for (int s = 128; s > 0; s >>= 1) { if (j < s) red[j] += red[j + s]; __syncthreads(); }
    P[row * cL + j] = (bf16)(e / red[0]);
}

__global__ __launch_bounds__(256)
void softmax_col(const float* __restrict__ G, bf16* __restrict__ P) // P[b][j][i]
{
    __shared__ float red[cL];
    const int i = threadIdx.x;
    const int j = blockIdx.x % cL;
    const int b = blockIdx.x / cL;
    const float v = G[((long)(b * cL + i)) * cL + j];
    red[i] = v; __syncthreads();
    for (int s = 128; s > 0; s >>= 1) { if (i < s) red[i] = fmaxf(red[i], red[i + s]); __syncthreads(); }
    const float mx = red[0]; __syncthreads();
    const float e = expf(v - mx);
    red[i] = e; __syncthreads();
    for (int s = 128; s > 0; s >>= 1) { if (i < s) red[i] += red[i + s]; __syncthreads(); }
    P[((long)(b * cL + j)) * cL + i] = (bf16)(e / red[0]);
}

// ============================================================== host dispatch
extern "C" void kernel_launch(void* const* d_in, const int* in_sizes, int n_in,
                              void* d_out, int out_size, void* d_ws, size_t ws_size,
                              hipStream_t stream)
{
    (void)in_sizes; (void)n_in; (void)out_size; (void)ws_size;
    const float* const* in = (const float* const*)d_in;

    char*  ws  = (char*)d_ws;
    size_t off = 0;
    auto alloc = [&](size_t bytes) -> void* {
        void* p = ws + off;
        off += (bytes + 255) & ~(size_t)255;
        return p;
    };

    const long nPix = (long)cB * cL * cL;                 // 131072
    const long nTok = (long)cB * cL;                      // 512

    bf16*  bigN = (bf16*)alloc(nPix * cD2 * 2);
    bf16*  bigC = (bf16*)alloc(nPix * cD2 * 2);
    bf16*  Xbf  = (bf16*)alloc(nTok * cD1 * 2);
    bf16*  H1   = (bf16*)alloc(nTok * cD1 * 2);
    bf16*  Bs0  = (bf16*)alloc(nTok * cD1 * 2);
    bf16*  Be0  = (bf16*)alloc(nTok * cD1 * 2);
    bf16*  Vt   = (bf16*)alloc(nTok * cD1 * 2);   // V transposed per batch [768][256]
    bf16*  Bsb  = (bf16*)alloc(nTok * cD1 * 2);
    bf16*  Beb  = (bf16*)alloc(nTok * cD1 * 2);
    float* Ps   = (float*)alloc(nTok * cD2 * 4);
    float* Pe   = (float*)alloc(nTok * cD2 * 4);
    float* G    = (float*)alloc(nPix * 4);
    bf16*  P    = (bf16*)alloc(nPix * 2);

    bf16* wT_besW1 = (bf16*)alloc(cD1 * cD1 * 2);
    bf16* wT_besW2 = (bf16*)alloc(cD1 * cD1 * 2);
    bf16* wT_beeW1 = (bf16*)alloc(cD1 * cD1 * 2);
    bf16* wT_beeW2 = (bf16*)alloc(cD1 * cD1 * 2);
    bf16* wT_W2s   = (bf16*)alloc(cD1 * cD1 * 2);
    bf16* wT_W2e   = (bf16*)alloc(cD1 * cD1 * 2);
    bf16* wT_WsS   = (bf16*)alloc(cD1 * cD2 * 2);
    bf16* wT_WsE   = (bf16*)alloc(cD1 * cD2 * 2);
    bf16* wT_Ws    = (bf16*)alloc(cD1 * cD2 * 2);
    bf16* wT_Ks    = (bf16*)alloc(25 * cD2 * cD2 * 2);
    bf16* wT_Ke    = (bf16*)alloc(25 * cD2 * cD2 * 2);
    bf16* wT_K     = (bf16*)alloc(25 * cD2 * cD2 * 2);
    bf16* wT_sW1s  = (bf16*)alloc(cD2 * cD2 * 2);
    bf16* wT_sW1e  = (bf16*)alloc(cD2 * cD2 * 2);
    bf16* wT_sW1   = (bf16*)alloc(cD2 * cD2 * 2);

    auto cvt = [&](const float* s, bf16* d, long n) {
        cvt_f32_bf16<<<dim3((unsigned)((n + 255) / 256)), dim3(256), 0, stream>>>(s, d, (int)n);
    };
    auto cvtT = [&](const float* s, bf16* d, int K, int N) {
        long n = (long)K * N;
        cvt_f32_bf16_t<<<dim3((unsigned)((n + 255) / 256)), dim3(256), 0, stream>>>(s, d, K, N);
    };
    auto gemm = [&](const bf16* A, const bf16* BT, const float* bias,
                    float* Cf, bf16* Cb, int M, int N, int K, int relu, int transM,
                    int batch, long sA, long sB, long sC) {
        gemm_bf16<<<dim3(M / 64, N / 64, batch), dim3(128), 0, stream>>>(
            A, BT, bias, Cf, Cb, M, N, K, relu, transM, sA, sB, sC);
    };

    const float* H_c   = in[0];
    const float* masks = in[2];
    // be_s:3..6  be_e:7..10 | se_s: Ws=11 bs=12 emb=13 g=14 b=15 | se_e:16..20
    // se:21..25 | conv_s: K=26 b=27 | conv_e:28,29 | conv:30,31
    // sc_s:32..35 sc_e:36..39 sc:40..43 | W2s:44,45 W2e:46,47

    cvt(H_c, Xbf, nTok * cD1);
    cvtT(in[3],  wT_besW1, cD1, cD1);  cvtT(in[5],  wT_besW2, cD1, cD1);
    cvtT(in[7],  wT_beeW1, cD1, cD1);  cvtT(in[9],  wT_beeW2, cD1, cD1);
    cvtT(in[44], wT_W2s,   cD1, cD1);  cvtT(in[46], wT_W2e,   cD1, cD1);
    cvtT(in[11], wT_WsS,   cD1, cD2);  cvtT(in[16], wT_WsE,   cD1, cD2);
    cvtT(in[21], wT_Ws,    cD1, cD2);
    cvtT(in[32], wT_sW1s,  cD2, cD2);  cvtT(in[36], wT_sW1e,  cD2, cD2);
    cvtT(in[40], wT_sW1,   cD2, cD2);
    {
        long n = 25L * cD2 * cD2;
        cvt_convw_t<<<dim3((unsigned)((n + 255) / 256)), dim3(256), 0, stream>>>(in[26], wT_Ks);
        cvt_convw_t<<<dim3((unsigned)((n + 255) / 256)), dim3(256), 0, stream>>>(in[28], wT_Ke);
        cvt_convw_t<<<dim3((unsigned)((n + 255) / 256)), dim3(256), 0, stream>>>(in[30], wT_K);
    }

    const int M  = (int)nTok;         // 512
    const int MP = (int)nPix;         // 131072
    const long sP = (long)cL * cL;    // 65536
    const long sV = (long)cL * cD1;   // 196608

    // ---- boundary MLPs ----
    gemm(Xbf, wT_besW1, in[4],  nullptr, H1,  M, cD1, cD1, 1, 0, 1, 0, 0, 0);
    gemm(H1,  wT_besW2, in[6],  nullptr, Bs0, M, cD1, cD1, 0, 0, 1, 0, 0, 0);
    gemm(Xbf, wT_beeW1, in[8],  nullptr, H1,  M, cD1, cD1, 1, 0, 1, 0, 0, 0);
    gemm(H1,  wT_beeW2, in[10], nullptr, Be0, M, cD1, cD1, 0, 0, 1, 0, 0, 0);

    // ---- start-boundary aggregation ----
    gemm(Bs0, wT_WsS, in[12], Ps, nullptr, M, cD2, cD1, 0, 0, 1, 0, 0, 0);
    gemm(Be0, wT_WsS, in[12], Pe, nullptr, M, cD2, cD1, 0, 0, 1, 0, 0, 0);
    span_ln<<<dim3(cL, cL, cB), dim3(cD2), 0, stream>>>(Ps, Pe, in[13], in[14], in[15], masks, 0, bigN);
    conv5x5_bf16<<<dim3(cL / 64, cL, cB), dim3(128), 0, stream>>>(bigN, wT_Ks, in[27], bigC);
    gemm(bigC, wT_sW1s, in[33], nullptr, bigN, MP, cD2, cD2, 1, 0, 1, 0, 0, 0);
    score_dot<<<dim3((unsigned)(nPix / 4)), dim3(128), 0, stream>>>(bigN, in[34], in[35], masks, G);
    softmax_row<<<dim3((unsigned)nTok), dim3(cL), 0, stream>>>(G, P);
    gemm(Bs0, wT_W2s, in[45], nullptr, Vt, M, cD1, cD1, 0, /*transM=*/cL, 1, 0, 0, sV);
    gemm(P, Vt, nullptr, nullptr, Bsb, cL, cD1, cL, 0, 0, cB, sP, sV, sV);

    // ---- end-boundary aggregation ----
    gemm(Bs0, wT_WsE, in[17], Ps, nullptr, M, cD2, cD1, 0, 0, 1, 0, 0, 0);
    gemm(Be0, wT_WsE, in[17], Pe, nullptr, M, cD2, cD1, 0, 0, 1, 0, 0, 0);
    span_ln<<<dim3(cL, cL, cB), dim3(cD2), 0, stream>>>(Ps, Pe, in[18], in[19], in[20], masks, 0, bigN);
    conv5x5_bf16<<<dim3(cL / 64, cL, cB), dim3(128), 0, stream>>>(bigN, wT_Ke, in[29], bigC);
    gemm(bigC, wT_sW1e, in[37], nullptr, bigN, MP, cD2, cD2, 1, 0, 1, 0, 0, 0);
    score_dot<<<dim3((unsigned)(nPix / 4)), dim3(128), 0, stream>>>(bigN, in[38], in[39], masks, G);
    softmax_col<<<dim3((unsigned)nTok), dim3(cL), 0, stream>>>(G, P);   // P[b][j][i]
    gemm(Be0, wT_W2e, in[47], nullptr, Vt, M, cD1, cD1, 0, /*transM=*/cL, 1, 0, 0, sV);
    gemm(P, Vt, nullptr, nullptr, Beb, cL, cD1, cL, 0, 0, cB, sP, sV, sV);

    // ---- final span representation + scoring ----
    gemm(Bsb, wT_Ws, in[22], Ps, nullptr, M, cD2, cD1, 0, 0, 1, 0, 0, 0);
    gemm(Beb, wT_Ws, in[22], Pe, nullptr, M, cD2, cD1, 0, 0, 1, 0, 0, 0);
    span_ln<<<dim3(cL, cL, cB), dim3(cD2), 0, stream>>>(Ps, Pe, in[23], in[24], in[25], masks, 1, bigN);
    conv5x5_bf16<<<dim3(cL / 64, cL, cB), dim3(128), 0, stream>>>(bigN, wT_K, in[31], bigC);
    gemm(bigC, wT_sW1, in[41], nullptr, bigN, MP, cD2, cD2, 1, 0, 1, 0, 0, 0);
    score_dot<<<dim3((unsigned)(nPix / 4)), dim3(128), 0, stream>>>(bigN, in[42], in[43], masks, (float*)d_out);
}
